// MetaPartModule_19396072309310
// MI455X (gfx1250) — matrix operand order repaired and checked by reading, not compiled, not run on hardware
//
#include <hip/hip_runtime.h>

typedef __attribute__((ext_vector_type(16))) _Float16 v16h;
typedef __attribute__((ext_vector_type(8)))  float    v8f;

constexpr int Bn = 64;      // batch
constexpr int Cc = 512;     // channels = d
constexpr int HW = 4096;    // h*w
constexpr int Nn = 32;      // embed_num
constexpr float BN_EPS = 1e-5f;
constexpr float EPS    = 1e-12f;

union HFrag { uint4 u[2]; v16h v; };

__device__ __forceinline__ v8f wmma_f16(v16h a, v16h b, v8f c) {
  return __builtin_amdgcn_wmma_f32_16x16x32_f16(false, a, false, b, (short)0, c, false, false);
}

// A fragment from f16 row-major source: lane holds M=lane%16 row, K runs
// kc*32+16g+0..15 contiguous (g = lane/16). 16B-aligned loads.
__device__ __forceinline__ v16h load_a_h16(const _Float16* row_ptr, int kc, int g) {
  HFrag f;
  const uint4* p = (const uint4*)(row_ptr + kc * 32 + 16 * g);
  f.u[0] = p[0];   // 8 halfs
  f.u[1] = p[1];   // 8 halfs at +8 elements
  return f.v;
}

// B fragment from f16 row-major [N][K] source (K contiguous): lane N=lane%16,
// K = kc*32 + 16g + 0..15 contiguous -> two b128 loads.
__device__ __forceinline__ v16h load_b_h16(const _Float16* n_row_ptr, int kc, int g) {
  HFrag f;
  const uint4* p = (const uint4*)(n_row_ptr + kc * 32 + 16 * g);
  f.u[0] = p[0];
  f.u[1] = p[1];
  return f.v;
}

// A fragment converted from fp32 row-major source (same K runs as load_a_h16).
__device__ __forceinline__ v16h cvt_a_f32(const float* row_ptr, int kc, int g) {
  HFrag f;
  const float* p1 = row_ptr + kc * 32 + 16 * g;
#pragma unroll
  for (int e = 0; e < 8; ++e) f.v[e] = (_Float16)p1[e];
#pragma unroll
  for (int e = 0; e < 8; ++e) f.v[8 + e] = (_Float16)p1[8 + e];
  return f.v;
}

// B fragment converted from fp32 row-major [N][K] source (K contiguous).
__device__ __forceinline__ v16h cvt_b_nrow_f32(const float* n_row_ptr, int kc, int g) {
  HFrag f;
  const float* p = n_row_ptr + kc * 32 + 16 * g;
#pragma unroll
  for (int e = 0; e < 16; ++e) f.v[e] = (_Float16)p[e];
  return f.v;
}

// B fragment converted from fp32 row-major [K][N] source (N contiguous): lane N
// column fixed, 16 K rows strided by ld.
__device__ __forceinline__ v16h cvt_b_krow_f32(const float* src, size_t ld, int kc, int g, int ncol) {
  HFrag f;
  const float* p = src + (size_t)(kc * 32 + 16 * g) * ld + ncol;
#pragma unroll
  for (int e = 0; e < 16; ++e) { f.v[e] = (_Float16)(*p); p += ld; }
  return f.v;
}

#define ACC_ZERO {{{0.f,0.f,0.f,0.f,0.f,0.f,0.f,0.f},{0.f,0.f,0.f,0.f,0.f,0.f,0.f,0.f}}, \
                  {{0.f,0.f,0.f,0.f,0.f,0.f,0.f,0.f},{0.f,0.f,0.f,0.f,0.f,0.f,0.f,0.f}}}

// ---------------- one-time convert: x -> xh[b][ch][p] f16 and xT[b][p][ch] f16 ----
__global__ void k_convert(const float* __restrict__ x, _Float16* __restrict__ xh,
                          _Float16* __restrict__ xT) {
  int b  = blockIdx.z;
  int p0 = blockIdx.x * 32, ch0 = blockIdx.y * 32;
  const float* xb = x + (size_t)b * Cc * HW;
  __shared__ float tile[32][33];
  int c = threadIdx.x & 31, r = threadIdx.x >> 5;   // r = 0..7
#pragma unroll
  for (int k = 0; k < 4; ++k)
    tile[r + 8 * k][c] = xb[(size_t)(ch0 + r + 8 * k) * HW + p0 + c];
  __syncthreads();
  _Float16* xhb = xh + (size_t)b * Cc * HW;
  _Float16* xTb = xT + (size_t)b * HW * Cc;
#pragma unroll
  for (int k = 0; k < 4; ++k) {
    xhb[(size_t)(ch0 + r + 8 * k) * HW + p0 + c] = (_Float16)tile[r + 8 * k][c];
    xTb[(size_t)(p0 + r + 8 * k) * Cc + ch0 + c] = (_Float16)tile[c][r + 8 * k];
  }
}

// ---------------- elementwise / init ----------------
__global__ void k_slots_init(const float* __restrict__ meta, float* __restrict__ slots) {
  int row = blockIdx.x;             // 0..2047 (b*32+i)
  int i = row & (Nn - 1);
  for (int t = threadIdx.x; t < Cc; t += 256)
    slots[(size_t)row * Cc + t] = meta[(size_t)i * Cc + t];
}

__global__ void k_blend(const float* __restrict__ ns, float* __restrict__ slots) {
  size_t idx = (size_t)blockIdx.x * 256 + threadIdx.x;
  slots[idx] = 0.1f * ns[idx] + 0.9f * slots[idx];
}

// ---------------- q projection: qlin = slots @ Wq^T + bq ----------------
__global__ void k_qlin(const float* __restrict__ slots, const float* __restrict__ Wq,
                       const float* __restrict__ bq, float* __restrict__ qlin) {
  int rBase  = blockIdx.y * 32;                 // 64 row-blocks
  int oBlock = blockIdx.x * 256;                // 2 col-blocks
  int wave = threadIdx.x >> 5, lane = threadIdx.x & 31;
  int g = lane >> 4, l16 = lane & 15;
  int oWave = oBlock + wave * 32;

  v8f acc[2][2] = ACC_ZERO;
  for (int kc = 0; kc < Cc / 32; ++kc) {
    v16h a[2], b[2];
#pragma unroll
    for (int mt = 0; mt < 2; ++mt)
      a[mt] = cvt_a_f32(slots + (size_t)(rBase + mt * 16 + l16) * Cc, kc, g);
#pragma unroll
    for (int nt = 0; nt < 2; ++nt)
      b[nt] = cvt_b_nrow_f32(Wq + (size_t)(oWave + nt * 16 + l16) * Cc, kc, g);
#pragma unroll
    for (int mt = 0; mt < 2; ++mt)
#pragma unroll
      for (int nt = 0; nt < 2; ++nt)
        acc[mt][nt] = wmma_f16(a[mt], b[nt], acc[mt][nt]);
  }
#pragma unroll
  for (int mt = 0; mt < 2; ++mt)
#pragma unroll
    for (int nt = 0; nt < 2; ++nt) {
      int col = oWave + nt * 16 + l16;
      float bias = bq[col];
#pragma unroll
      for (int r = 0; r < 8; ++r) {
        int M = rBase + mt * 16 + 8 * g + r;
        qlin[(size_t)M * Cc + col] = acc[mt][nt][r] + bias;
      }
    }
}

// per-column sum / sumsq of qlin (2048 x 512) -> stats[0..511]=sum, [512..1023]=sumsq
__global__ void k_colstats(const float* __restrict__ qlin, float* __restrict__ stats) {
  int c0 = blockIdx.x * 32;           // 16 col groups
  int r0 = blockIdx.y * 256;          // 8 row groups
  int c  = threadIdx.x & 31;
  int rg = threadIdx.x >> 5;          // 0..7
  float s = 0.f, s2 = 0.f;
  for (int r = r0 + rg; r < r0 + 256; r += 8) {
    float v = qlin[(size_t)r * Cc + c0 + c];
    s += v; s2 += v * v;
  }
  __shared__ float ls[8][32], ls2[8][32];
  ls[rg][c] = s; ls2[rg][c] = s2;
  __syncthreads();
  if (threadIdx.x < 32) {
    float t = 0.f, t2 = 0.f;
#pragma unroll
    for (int j = 0; j < 8; ++j) { t += ls[j][threadIdx.x]; t2 += ls2[j][threadIdx.x]; }
    atomicAdd(&stats[c0 + threadIdx.x], t);
    atomicAdd(&stats[Cc + c0 + threadIdx.x], t2);
  }
}

__global__ void k_qfinal(const float* __restrict__ qlin, const float* __restrict__ stats,
                         const float* __restrict__ g1, const float* __restrict__ b1,
                         _Float16* __restrict__ qh) {
  size_t idx = (size_t)blockIdx.x * 256 + threadIdx.x;   // 1M elements
  int c = (int)(idx & (Cc - 1));
  float mean = stats[c] * (1.f / 2048.f);
  float var  = stats[Cc + c] * (1.f / 2048.f) - mean * mean;
  float v = (qlin[idx] - mean) * rsqrtf(var + BN_EPS) * g1[c] + b1[c];
  qh[idx] = (_Float16)fmaxf(v, 0.f);
}

// ---------------- attn = q @ k^T   (per batch: 32x512 @ 512x4096) ----------------
// fast path: B from xT[b][p][ch] f16 (K contiguous -> b128 loads)
__global__ void k_attn_f16(const _Float16* __restrict__ xT, const _Float16* __restrict__ qh,
                           float* __restrict__ attn) {
  int b = blockIdx.y;
  int pBlock = blockIdx.x * 256;
  int wave = threadIdx.x >> 5, lane = threadIdx.x & 31;
  int g = lane >> 4, l16 = lane & 15;
  int pWave = pBlock + wave * 32;
  const _Float16* xTb = xT + (size_t)b * HW * Cc;
  const _Float16* qb  = qh + (size_t)b * Nn * Cc;

  v8f acc[2][2] = ACC_ZERO;
  for (int kc = 0; kc < Cc / 32; ++kc) {
    v16h a[2], bf[2];
#pragma unroll
    for (int mt = 0; mt < 2; ++mt)
      a[mt] = load_a_h16(qb + (size_t)(mt * 16 + l16) * Cc, kc, g);
#pragma unroll
    for (int nt = 0; nt < 2; ++nt)
      bf[nt] = load_b_h16(xTb + (size_t)(pWave + nt * 16 + l16) * Cc, kc, g);
#pragma unroll
    for (int mt = 0; mt < 2; ++mt)
#pragma unroll
      for (int nt = 0; nt < 2; ++nt)
        acc[mt][nt] = wmma_f16(a[mt], bf[nt], acc[mt][nt]);
  }
  float* ab = attn + (size_t)b * Nn * HW;
#pragma unroll
  for (int mt = 0; mt < 2; ++mt)
#pragma unroll
    for (int nt = 0; nt < 2; ++nt) {
      int col = pWave + nt * 16 + l16;
#pragma unroll
      for (int r = 0; r < 8; ++r)
        ab[(size_t)(mt * 16 + 8 * g + r) * HW + col] = acc[mt][nt][r];
    }
}

// fallback: B gathered from fp32 x[b][ch][p]
__global__ void k_attn_f32(const float* __restrict__ x, const _Float16* __restrict__ qh,
                           float* __restrict__ attn) {
  int b = blockIdx.y;
  int pBlock = blockIdx.x * 256;
  int wave = threadIdx.x >> 5, lane = threadIdx.x & 31;
  int g = lane >> 4, l16 = lane & 15;
  int pWave = pBlock + wave * 32;
  const float*    xb = x  + (size_t)b * Cc * HW;
  const _Float16* qb = qh + (size_t)b * Nn * Cc;

  v8f acc[2][2] = ACC_ZERO;
  for (int kc = 0; kc < Cc / 32; ++kc) {
    v16h a[2], bf[2];
#pragma unroll
    for (int mt = 0; mt < 2; ++mt)
      a[mt] = load_a_h16(qb + (size_t)(mt * 16 + l16) * Cc, kc, g);
#pragma unroll
    for (int nt = 0; nt < 2; ++nt)
      bf[nt] = cvt_b_krow_f32(xb, (size_t)HW, kc, g, pWave + nt * 16 + l16);
#pragma unroll
    for (int mt = 0; mt < 2; ++mt)
#pragma unroll
      for (int nt = 0; nt < 2; ++nt)
        acc[mt][nt] = wmma_f16(a[mt], bf[nt], acc[mt][nt]);
  }
  float* ab = attn + (size_t)b * Nn * HW;
#pragma unroll
  for (int mt = 0; mt < 2; ++mt)
#pragma unroll
    for (int nt = 0; nt < 2; ++nt) {
      int col = pWave + nt * 16 + l16;
#pragma unroll
      for (int r = 0; r < 8; ++r)
        ab[(size_t)(mt * 16 + 8 * g + r) * HW + col] = acc[mt][nt][r];
    }
}

// per-channel (n) stats over (b, hw): one block per (b,i) row, atomics into astats[64]
__global__ void k_attnstats(const float* __restrict__ attn, float* __restrict__ astats) {
  int row = blockIdx.x;              // b*32+i
  int i = row & (Nn - 1);
  const float* p = attn + (size_t)row * HW;
  float s = 0.f, s2 = 0.f;
  for (int t = threadIdx.x; t < HW; t += 256) { float v = p[t]; s += v; s2 += v * v; }
  __shared__ float ss[256], ss2[256];
  ss[threadIdx.x] = s; ss2[threadIdx.x] = s2;
  __syncthreads();
  for (int off = 128; off > 0; off >>= 1) {
    if (threadIdx.x < off) { ss[threadIdx.x] += ss[threadIdx.x + off]; ss2[threadIdx.x] += ss2[threadIdx.x + off]; }
    __syncthreads();
  }
  if (threadIdx.x == 0) { atomicAdd(&astats[i], ss[0]); atomicAdd(&astats[Nn + i], ss2[0]); }
}

// normalize+relu attn -> f16, and per-row sum
__global__ void k_attnfinal(const float* __restrict__ attn, const float* __restrict__ astats,
                            const float* __restrict__ g2, const float* __restrict__ b2,
                            _Float16* __restrict__ ah, float* __restrict__ rowsum) {
  int row = blockIdx.x;
  int i = row & (Nn - 1);
  const float cnt = 1.f / (float)(Bn * HW);
  float mean = astats[i] * cnt;
  float var  = astats[Nn + i] * cnt - mean * mean;
  float scale = rsqrtf(var + BN_EPS) * g2[i];
  float shift = b2[i] - mean * scale;
  const float* p = attn + (size_t)row * HW;
  _Float16* o = ah + (size_t)row * HW;
  float s = 0.f;
  for (int t = threadIdx.x; t < HW; t += 256) {
    float v = fmaxf(p[t] * scale + shift, 0.f);
    o[t] = (_Float16)v;
    s += v;
  }
  __shared__ float ss[256];
  ss[threadIdx.x] = s;
  __syncthreads();
  for (int off = 128; off > 0; off >>= 1) {
    if (threadIdx.x < off) ss[threadIdx.x] += ss[threadIdx.x + off];
    __syncthreads();
  }
  if (threadIdx.x == 0) rowsum[row] = ss[0];
}

// ---------------- out = attn' @ v / rowsum  (per batch: 32x4096 @ 4096x512) --------
// fast path: B from xh[b][ch][p] f16 (K=p contiguous -> b128 loads)
__global__ void k_out_f16(const _Float16* __restrict__ xh, const _Float16* __restrict__ ah,
                          const float* __restrict__ rowsum, float* __restrict__ dst) {
  int b = blockIdx.y;
  int chBlock = blockIdx.x * 256;
  int wave = threadIdx.x >> 5, lane = threadIdx.x & 31;
  int g = lane >> 4, l16 = lane & 15;
  int chWave = chBlock + wave * 32;
  const _Float16* xhb = xh + (size_t)b * Cc * HW;
  const _Float16* ab  = ah + (size_t)b * Nn * HW;

  v8f acc[2][2] = ACC_ZERO;
  for (int kc = 0; kc < HW / 32; ++kc) {
    v16h a[2], bf[2];
#pragma unroll
    for (int mt = 0; mt < 2; ++mt)
      a[mt] = load_a_h16(ab + (size_t)(mt * 16 + l16) * HW, kc, g);
#pragma unroll
    for (int nt = 0; nt < 2; ++nt)
      bf[nt] = load_b_h16(xhb + (size_t)(chWave + nt * 16 + l16) * HW, kc, g);
#pragma unroll
    for (int mt = 0; mt < 2; ++mt)
#pragma unroll
      for (int nt = 0; nt < 2; ++nt)
        acc[mt][nt] = wmma_f16(a[mt], bf[nt], acc[mt][nt]);
  }
#pragma unroll
  for (int mt = 0; mt < 2; ++mt)
#pragma unroll
    for (int nt = 0; nt < 2; ++nt) {
      int col = chWave + nt * 16 + l16;
#pragma unroll
      for (int r = 0; r < 8; ++r) {
        int M = mt * 16 + 8 * g + r;
        float inv = 1.f / (rowsum[b * Nn + M] + EPS);
        dst[(size_t)(b * Nn + M) * Cc + col] = acc[mt][nt][r] * inv;
      }
    }
}

// fallback: B converted from fp32 x
__global__ void k_out_f32(const float* __restrict__ x, const _Float16* __restrict__ ah,
                          const float* __restrict__ rowsum, float* __restrict__ dst) {
  int b = blockIdx.y;
  int chBlock = blockIdx.x * 256;
  int wave = threadIdx.x >> 5, lane = threadIdx.x & 31;
  int g = lane >> 4, l16 = lane & 15;
  int chWave = chBlock + wave * 32;
  const float*    xb = x  + (size_t)b * Cc * HW;
  const _Float16* ab = ah + (size_t)b * Nn * HW;

  v8f acc[2][2] = ACC_ZERO;
  for (int kc = 0; kc < HW / 32; ++kc) {
    v16h a[2], bf[2];
#pragma unroll
    for (int mt = 0; mt < 2; ++mt)
      a[mt] = load_a_h16(ab + (size_t)(mt * 16 + l16) * HW, kc, g);
#pragma unroll
    for (int nt = 0; nt < 2; ++nt)
      bf[nt] = cvt_b_nrow_f32(xb + (size_t)(chWave + nt * 16 + l16) * HW, kc, g);
#pragma unroll
    for (int mt = 0; mt < 2; ++mt)
#pragma unroll
      for (int nt = 0; nt < 2; ++nt)
        acc[mt][nt] = wmma_f16(a[mt], bf[nt], acc[mt][nt]);
  }
#pragma unroll
  for (int mt = 0; mt < 2; ++mt)
#pragma unroll
    for (int nt = 0; nt < 2; ++nt) {
      int col = chWave + nt * 16 + l16;
#pragma unroll
      for (int r = 0; r < 8; ++r) {
        int M = mt * 16 + 8 * g + r;
        float inv = 1.f / (rowsum[b * Nn + M] + EPS);
        dst[(size_t)(b * Nn + M) * Cc + col] = acc[mt][nt][r] * inv;
      }
    }
}

extern "C" void kernel_launch(void* const* d_in, const int* in_sizes, int n_in,
                              void* d_out, int out_size, void* d_ws, size_t ws_size,
                              hipStream_t stream) {
  (void)in_sizes; (void)n_in; (void)out_size;
  const float* x    = (const float*)d_in[0];
  const float* meta = (const float*)d_in[1];
  const float* Wq   = (const float*)d_in[2];
  const float* bq   = (const float*)d_in[3];
  const float* g1   = (const float*)d_in[4];
  const float* b1   = (const float*)d_in[5];
  const float* g2   = (const float*)d_in[6];
  const float* b2   = (const float*)d_in[7];
  float* out = (float*)d_out;

  char* w = (char*)d_ws;
  auto take = [&](size_t bytes) { void* p = (void*)w; w += (bytes + 255) & ~(size_t)255; return p; };
  float*    slots  = (float*)   take((size_t)2048 * Cc * 4);     // 4 MB
  float*    ns     = (float*)   take((size_t)2048 * Cc * 4);     // 4 MB
  float*    qlin   = (float*)   take((size_t)2048 * Cc * 4);     // 4 MB
  _Float16* qh     = (_Float16*)take((size_t)2048 * Cc * 2);     // 2 MB
  float*    attn   = (float*)   take((size_t)Bn * Nn * HW * 4);  // 32 MB
  _Float16* ah     = (_Float16*)take((size_t)Bn * Nn * HW * 2);  // 16 MB
  float*    qstats = (float*)   take(2 * Cc * 4);
  float*    astats = (float*)   take(2 * Nn * 4);
  float*    rowsum = (float*)   take(2048 * 4);

  // Fast path: two one-time f16 copies of x (xh: [ch][p], xT: [p][ch]); 512 MB.
  const size_t xcopy_bytes = (size_t)Bn * Cc * HW * 2;
  size_t used = (size_t)(w - (char*)d_ws);
  bool useF16 = (ws_size > used) && ((ws_size - used) >= 2 * xcopy_bytes + 1024);
  _Float16* xh = nullptr;
  _Float16* xT = nullptr;
  if (useF16) {
    xh = (_Float16*)take(xcopy_bytes);
    xT = (_Float16*)take(xcopy_bytes);
    k_convert<<<dim3(HW / 32, Cc / 32, Bn), 256, 0, stream>>>(x, xh, xT);
  }

  k_slots_init<<<2048, 256, 0, stream>>>(meta, slots);
  for (int it = 0; it < 3; ++it) {
    if (it > 0) k_blend<<<4096, 256, 0, stream>>>(ns, slots);
    hipMemsetAsync(qstats, 0, 2 * Cc * 4, stream);
    k_qlin<<<dim3(2, 64), 256, 0, stream>>>(slots, Wq, bq, qlin);
    k_colstats<<<dim3(16, 8), 256, 0, stream>>>(qlin, qstats);
    k_qfinal<<<4096, 256, 0, stream>>>(qlin, qstats, g1, b1, qh);
    hipMemsetAsync(astats, 0, 2 * Nn * 4, stream);
    if (useF16) k_attn_f16<<<dim3(16, 64), 256, 0, stream>>>(xT, qh, attn);
    else        k_attn_f32<<<dim3(16, 64), 256, 0, stream>>>(x, qh, attn);
    k_attnstats<<<2048, 256, 0, stream>>>(attn, astats);
    k_attnfinal<<<2048, 256, 0, stream>>>(attn, astats, g2, b2, ah, rowsum);
    float* dst = (it == 2) ? out : ns;
    if (useF16) k_out_f16<<<dim3(2, 64), 256, 0, stream>>>(xh, ah, rowsum, dst);
    else        k_out_f32<<<dim3(2, 64), 256, 0, stream>>>(x, ah, rowsum, dst);
  }
}
